// LigerLMHeadCE_3100966388353
// MI455X (gfx1250) — matrix operand order repaired and checked by reading, not compiled, not run on hardware
//
#include <hip/hip_runtime.h>
#include <math.h>

#define IGNORE_INDEX (-100)
#define N_TOK 4096
#define H_DIM 4096
#define V_DIM 32000

#define WAVES 8
#define LDS_STRIDE 4104              // 4096 bf16 + 8 pad to spread LDS banks
#define NUM_V_ITERS (V_DIM / (WAVES * 16))   // 250

typedef __attribute__((ext_vector_type(16))) __bf16    v16bf;
typedef __attribute__((ext_vector_type(8)))  float     v8f;
typedef __attribute__((ext_vector_type(8)))  unsigned  v8u;

// one-time staging conversion: round-to-nearest-even fp32 -> bf16 bits
__device__ __forceinline__ unsigned short f2bf(float a) {
  unsigned u = __float_as_uint(a);
  u += 0x7FFFu + ((u >> 16) & 1u);
  return (unsigned short)(u >> 16);
}
// hot-path pack: two fp32 -> two bf16 (truncate) in ONE v_perm_b32
// result bytes [1:0] = bytes [3:2] of a (low bf16), [3:2] = bytes [3:2] of b
__device__ __forceinline__ unsigned pack_bf16x2(float a, float b) {
  return __builtin_amdgcn_perm(__float_as_uint(b), __float_as_uint(a),
                               0x07060302u);
}

__global__ __launch_bounds__(256) void lmhead_ce_main(
    const float* __restrict__ X, const int* __restrict__ Y,
    const float* __restrict__ W, float* __restrict__ ws) {
  __shared__ unsigned short lds_x[16 * LDS_STRIDE];   // 16 rows of x, bf16
  __shared__ float s_max[WAVES][16];
  __shared__ float s_sum[WAVES][16];
  __shared__ float s_tl[WAVES][16];
  __shared__ int   s_y[16];

  const int tid  = threadIdx.x;
  const int wave = tid >> 5;
  const int lane = tid & 31;
  const int half = lane >> 4;        // K-half selector (A/B layouts)
  const int l15  = lane & 15;
  const int row0 = blockIdx.x * 16;

  if (tid < 16) s_y[tid] = Y[row0 + tid];
  // Stage the 16 x-rows into LDS as bf16 (coalesced, one-time).
  for (int idx = tid; idx < 16 * H_DIM; idx += 256) {
    int r = idx >> 12;               // H_DIM == 4096
    int c = idx & (H_DIM - 1);
    lds_x[r * LDS_STRIDE + c] = f2bf(X[(size_t)(row0 + r) * H_DIM + c]);
  }
  __syncthreads();

  // per-lane running online-softmax state over the columns this lane owns;
  // row m = j + 8*half in C-layout terms
  float rm[8], rs[8], tl[8];
  int   yreg[8];
#pragma unroll
  for (int j = 0; j < 8; ++j) {
    rm[j] = -INFINITY; rs[j] = 0.0f; tl[j] = 0.0f;
    yreg[j] = s_y[j + (half << 3)];
  }

  for (int it = 0; it < NUM_V_ITERS; ++it) {
    const int v0   = (it * WAVES + wave) * 16;
    const int vcol = v0 + l15;       // the logit column this lane holds in C
    // B source: lane supplies column n = l15 -> W row (v0 + l15),
    // K-half picks columns [half*8, half*8+8) and [16+half*8, 24+half*8) within each K-step.
    const float* wrow = W + (size_t)vcol * H_DIM + (half << 3);
    const unsigned short* arow = &lds_x[l15 * LDS_STRIDE + (half << 3)];

    v8f c = {};
#pragma unroll 2
    for (int kk = 0; kk < H_DIM; kk += 32) {
      // A fragment (16-bit A 16x32 layout): two 16B LDS reads
      const uint4* pa = reinterpret_cast<const uint4*>(arow + kk);
      uint4 a0 = pa[0];              // K = kbase .. kbase+7
      uint4 a1 = pa[2];              // K = kbase+16 .. kbase+23
      v8u au = {a0.x, a0.y, a0.z, a0.w, a1.x, a1.y, a1.z, a1.w};

      // B fragment: two runs of 8 fp32 (K +0..7, +16..23) of one W row -> 8 v_perm packs
      const float4* pb = reinterpret_cast<const float4*>(wrow + kk);
      float4 b0 = pb[0], b1 = pb[1], b2 = pb[4], b3 = pb[5];
      __builtin_prefetch(wrow + kk + 512, 0, 1);   // stream-ahead hint
      v8u bu = { pack_bf16x2(b0.x, b0.y), pack_bf16x2(b0.z, b0.w),
                 pack_bf16x2(b1.x, b1.y), pack_bf16x2(b1.z, b1.w),
                 pack_bf16x2(b2.x, b2.y), pack_bf16x2(b2.z, b2.w),
                 pack_bf16x2(b3.x, b3.y), pack_bf16x2(b3.z, b3.w) };

      v16bf A = __builtin_bit_cast(v16bf, au);
      v16bf B = __builtin_bit_cast(v16bf, bu);
      c = __builtin_amdgcn_wmma_f32_16x16x32_bf16(
              false, A, false, B, (short)0, c, false, false);
    }

    // branchless per-lane epilogue: online-softmax update + target pluck
#pragma unroll
    for (int j = 0; j < 8; ++j) {
      float v  = c[j];
      float nm = fmaxf(rm[j], v);
      rs[j] = rs[j] * __expf(rm[j] - nm) + __expf(v - nm);
      rm[j] = nm;
      tl[j] += (yreg[j] == vcol) ? v : 0.0f;
    }
  }

  // ONE cross-lane merge at the end (xor offsets < 16 stay within a half,
  // which is exactly where row m's 16 columns live)
#pragma unroll
  for (int j = 0; j < 8; ++j) {
    float m = rm[j], s = rs[j], t = tl[j];
#pragma unroll
    for (int off = 1; off < 16; off <<= 1) {
      float mo = __shfl_xor(m, off, 32);
      float so = __shfl_xor(s, off, 32);
      t += __shfl_xor(t, off, 32);
      float nm = fmaxf(m, mo);
      s = s * __expf(m - nm) + so * __expf(mo - nm);
      m = nm;
    }
    rm[j] = m; rs[j] = s; tl[j] = t;
  }

  // publish per-wave stats (values uniform within each half-wave)
  if (l15 == 0) {
#pragma unroll
    for (int j = 0; j < 8; ++j) {
      int rrow = j + (half << 3);
      s_max[wave][rrow] = rm[j];
      s_sum[wave][rrow] = rs[j];
      s_tl [wave][rrow] = tl[j];
    }
  }
  __syncthreads();

  // combine the 8 waves, emit per-row loss
  if (tid < 16) {
    float M = -INFINITY;
#pragma unroll
    for (int w = 0; w < WAVES; ++w) M = fmaxf(M, s_max[w][tid]);
    float S = 0.0f, T = 0.0f;
#pragma unroll
    for (int w = 0; w < WAVES; ++w) {
      S += s_sum[w][tid] * __expf(s_max[w][tid] - M);
      T += s_tl[w][tid];
    }
    float lse = M + __logf(S);
    int t = s_y[tid];
    if (t != IGNORE_INDEX) {
      atomicAdd(&ws[0], lse - T);
      atomicAdd(&ws[1], 1.0f);
    }
  }
}

__global__ void lmhead_ce_finalize(const float* __restrict__ ws,
                                   float* __restrict__ out) {
  out[0] = ws[0] / fmaxf(ws[1], 1.0f);
}

extern "C" void kernel_launch(void* const* d_in, const int* in_sizes, int n_in,
                              void* d_out, int out_size, void* d_ws, size_t ws_size,
                              hipStream_t stream) {
  (void)in_sizes; (void)n_in; (void)out_size; (void)ws_size;
  const float* X = (const float*)d_in[0];
  const int*   Y = (const int*)d_in[1];
  const float* W = (const float*)d_in[2];
  float* ws = (float*)d_ws;

  hipMemsetAsync(d_ws, 0, 2 * sizeof(float), stream);
  lmhead_ce_main<<<N_TOK / 16, 256, 0, stream>>>(X, Y, W, ws);
  lmhead_ce_finalize<<<1, 1, 0, stream>>>(ws, (float*)d_out);
}
